// ConvMod_Large_27513560498542
// MI455X (gfx1250) — matrix operand order repaired and checked by reading, not compiled, not run on hardware
//
#include <hip/hip_runtime.h>
#include <math.h>

// ---------------------------------------------------------------------------
// ConvMod_Large forward for MI455X (gfx1250, wave32, WMMA).
//
// Channels-last activations (NL x C, NL = B*L = 16384); every pointwise conv
// is a WMMA bf16 GEMM (fp32 accumulate) with fused epilogues. K is a template
// parameter (64/128/192) so the K-loop fully unrolls. NT = number of 16-col
// output strips per wave: NT=4 for the C in {64,128,192} GEMMs (one A fetch
// feeds 4 back-to-back v_wmma issues), NT=1 for the tiny offset/mask heads.
// OOB weight columns are loaded from a clamped row instead of being zeroed:
// output column n depends only on B column n, so garbage in unsaved columns
// is harmless and the load path stays branch-free (EXEC all ones through
// every WMMA, as the ISA requires).
// ---------------------------------------------------------------------------

#define BSZ   8
#define LSEQ  2048
#define DIMC  256
#define NLC   (BSZ * LSEQ)   // 16384 rows

typedef __bf16 v16bf __attribute__((ext_vector_type(16)));
typedef float  v8f   __attribute__((ext_vector_type(8)));

__device__ __forceinline__ float gelu_exact(float x) {
  return 0.5f * x * (1.0f + erff(x * 0.70710678118654752f));
}

// epilogue modes
#define EP_BIAS 0
#define EP_GELU 1
#define EP_MUL  2
#define EP_ADD  3
// store modes
#define ST_CL 0
#define ST_CF 1

__device__ __forceinline__ void cvt8(v16bf& d, int base, float4 x, float4 y) {
  d[base + 0] = (__bf16)x.x; d[base + 1] = (__bf16)x.y;
  d[base + 2] = (__bf16)x.z; d[base + 3] = (__bf16)x.w;
  d[base + 4] = (__bf16)y.x; d[base + 5] = (__bf16)y.y;
  d[base + 6] = (__bf16)y.z; d[base + 7] = (__bf16)y.w;
}

// ---------------------------------------------------------------------------
// WMMA GEMM: out = epilogue(A * W^T + bias). One wave -> 16 x (16*NT) tile.
// Lane layouts (A/B k order is this code's own, same for both; C/D per CDNA5 ISA 7.12.2):
//   A 16x32 bf16: lane.l16 = row M; element e -> K = (half<<4) + e
//     => per lane two adjacent 8-float runs: [half*16, +8) and [half*16+8, +8)
//   B 32x16 bf16: lane.l16 = col N; element e -> K = (half<<4) + e
//     => per lane one contiguous 16-float run: [half*16, +16)
//   C/D 16x16 f32: lane.l16 = col N; VGPR v -> row M = half*8 + v
// ---------------------------------------------------------------------------
template<int KD, int NT, int EP, int STM>
__global__ __launch_bounds__(128)
void k_gemm(const float* __restrict__ A, int lda,
            const float* __restrict__ W, const float* __restrict__ bias,
            int Md,
            const float* __restrict__ other, int ldo,
            float* __restrict__ out, int ldout, int colbase)
{
  const int lane = threadIdx.x & 31;
  const int wave = threadIdx.x >> 5;
  const int half = lane >> 4;
  const int l16  = lane & 15;
  const int row0 = (blockIdx.x * 4 + wave) * 16;       // exact tiling of NLC
  const int col0 = blockIdx.y * (16 * NT);

  const int arow = row0 + l16;
  const float* __restrict__ ap = A + (size_t)arow * lda;

  const float* wp[NT];
#pragma unroll
  for (int t = 0; t < NT; ++t) {
    int bc = col0 + t * 16 + l16;
    if (bc >= Md) bc = Md - 1;        // clamp: loads stay in-bounds, garbage
    wp[t] = W + (size_t)bc * KD;      // columns are never stored
  }

  v8f acc[NT];
#pragma unroll
  for (int t = 0; t < NT; ++t)
    acc[t] = (v8f){0.f, 0.f, 0.f, 0.f, 0.f, 0.f, 0.f, 0.f};

#pragma unroll
  for (int k0 = 0; k0 < KD; k0 += 32) {
    const float4* __restrict__ a0 = (const float4*)(ap + k0 + (half << 4));
    const float4* __restrict__ a1 = (const float4*)(ap + k0 + 8 + (half << 4));
    const float4 av0 = a0[0], av1 = a0[1];
    const float4 av2 = a1[0], av3 = a1[1];
    v16bf a;
    cvt8(a, 0, av0, av1);
    cvt8(a, 8, av2, av3);
#pragma unroll
    for (int t = 0; t < NT; ++t) {
      const float4* __restrict__ b0 = (const float4*)(wp[t] + k0 + (half << 4));
      const float4 bv0 = b0[0], bv1 = b0[1], bv2 = b0[2], bv3 = b0[3];
      v16bf b;
      cvt8(b, 0, bv0, bv1);
      cvt8(b, 8, bv2, bv3);
      acc[t] = __builtin_amdgcn_wmma_f32_16x16x32_bf16(false, a, false, b,
                                                       (short)0, acc[t],
                                                       false, false);
    }
  }

  // Epilogue (divergence only after all WMMAs have issued).
#pragma unroll
  for (int t = 0; t < NT; ++t) {
    const int ocol = col0 + t * 16 + l16;
    if (ocol >= Md) continue;
    const float bs = bias[ocol];
#pragma unroll
    for (int v = 0; v < 8; ++v) {
      const int orow = row0 + half * 8 + v;
      float x = acc[t][v] + bs;
      if (EP == EP_GELU)      x = gelu_exact(x);
      else if (EP == EP_MUL)  x *= other[(size_t)orow * ldo + ocol];
      else if (EP == EP_ADD)  x += other[(size_t)orow * ldo + ocol];
      if (STM == ST_CL) {
        out[(size_t)orow * ldout + ocol] = x;
      } else {
        const int n = orow >> 11, l = orow & (LSEQ - 1);
        out[((size_t)n * DIMC + colbase + ocol) * LSEQ + l] = x;
      }
    }
  }
}

// Channels-first LayerNorm of x(N,256,L) -> channels-last XN(NL,256).
__global__ void k_ln_cf2cl(const float* __restrict__ x,
                           const float* __restrict__ g, const float* __restrict__ b,
                           float* __restrict__ out)
{
  const int pos = blockIdx.x * blockDim.x + threadIdx.x;
  if (pos >= NLC) return;
  const int n = pos >> 11, l = pos & (LSEQ - 1);
  const float* xp = x + (size_t)n * DIMC * LSEQ + l;
  float s = 0.f, ss = 0.f;
  for (int c = 0; c < DIMC; ++c) {
    const float v = xp[(size_t)c * LSEQ];
    s += v; ss += v * v;
  }
  const float mu = s * (1.f / DIMC);
  const float var = ss * (1.f / DIMC) - mu * mu;
  const float r = rsqrtf(var + 1e-6f);
  for (int c = 0; c < DIMC; ++c) {
    const float v = (xp[(size_t)c * LSEQ] - mu) * r;
    out[(size_t)pos * DIMC + c] = g[c] * v + b[c];
  }
}

// Channels-last LayerNorm over C (compact rows).
__global__ void k_ln_cl(const float* __restrict__ in, int C,
                        const float* __restrict__ g, const float* __restrict__ b,
                        float* __restrict__ out)
{
  const int row = blockIdx.x * blockDim.x + threadIdx.x;
  if (row >= NLC) return;
  const float* p = in + (size_t)row * C;
  float s = 0.f, ss = 0.f;
  for (int c = 0; c < C; ++c) { const float v = p[c]; s += v; ss += v * v; }
  const float mu = s / C;
  const float var = ss / C - mu * mu;
  const float r = rsqrtf(var + 1e-6f);
  for (int c = 0; c < C; ++c)
    out[(size_t)row * C + c] = g[c] * ((p[c] - mu) * r) + b[c];
}

// Depthwise conv over L, channels-last. mode: 0=bias, 1=bias+GELU, 2=bias+add other.
__global__ void k_dwconv(const float* __restrict__ in, int ldin, int C,
                         const float* __restrict__ wgt, const float* __restrict__ bias,
                         int Kk, int mode, const float* __restrict__ other, int ldoth,
                         float* __restrict__ out, int ldout)
{
  const int idx = blockIdx.x * blockDim.x + threadIdx.x;
  if (idx >= NLC * C) return;
  const int c = idx % C, row = idx / C;
  const int n = row >> 11, l = row & (LSEQ - 1);
  const int pad = (Kk - 1) >> 1;
  const float* wc = wgt + (size_t)c * Kk;
  float s = bias[c];
  for (int k = 0; k < Kk; ++k) {
    const int ls = l + k - pad;
    if (ls >= 0 && ls < LSEQ)
      s += wc[k] * in[((size_t)n * LSEQ + ls) * ldin + c];
  }
  if (mode == 1)      s = gelu_exact(s);
  else if (mode == 2) s += other[(size_t)row * ldoth + c];
  out[(size_t)row * ldout + c] = s;
}

// In-place softmax over K for (NL, G, K) mask logits.
__global__ void k_softmax(float* __restrict__ m, int G, int Kk)
{
  const int i = blockIdx.x * blockDim.x + threadIdx.x;
  if (i >= NLC * G) return;
  float* p = m + (size_t)i * Kk;
  float mx = -3.4e38f;
  for (int k = 0; k < Kk; ++k) mx = fmaxf(mx, p[k]);
  float s = 0.f;
  for (int k = 0; k < Kk; ++k) { const float e = __expf(p[k] - mx); p[k] = e; s += e; }
  const float r = 1.f / s;
  for (int k = 0; k < Kk; ++k) p[k] *= r;
}

// DCNv3 1-D sampling: value(NL,C), offset/mask(NL,G*K) -> out(NL,C).
__global__ void k_dcn_sample(const float* __restrict__ val,
                             const float* __restrict__ off,
                             const float* __restrict__ msk,
                             int C, int G, int Kk,
                             float* __restrict__ out)
{
  const int idx = blockIdx.x * blockDim.x + threadIdx.x;
  if (idx >= NLC * C) return;
  const int c = idx % C, row = idx / C;
  const int n = row >> 11, l = row & (LSEQ - 1);
  const int Cg = C / G, g = c / Cg;
  const float* op = off + (size_t)row * (G * Kk) + g * Kk;
  const float* mp = msk + (size_t)row * (G * Kk) + g * Kk;
  const float cof = (Kk - 1) * 0.5f;
  float s = 0.f;
  for (int k = 0; k < Kk; ++k) {
    const float pos = (float)l + ((float)k - cof) + op[k];
    const float p0 = floorf(pos);
    const float fr = pos - p0;
    int i0 = (int)p0;      i0 = min(max(i0, 0), LSEQ - 1);
    int i1 = (int)p0 + 1;  i1 = min(max(i1, 0), LSEQ - 1);
    const float in0 = (p0 >= 0.f && p0 <= (float)(LSEQ - 1)) ? 1.f : 0.f;
    const float in1 = (p0 + 1.f >= 0.f && p0 + 1.f <= (float)(LSEQ - 1)) ? 1.f : 0.f;
    const float v0 = val[((size_t)n * LSEQ + i0) * C + c];
    const float v1 = val[((size_t)n * LSEQ + i1) * C + c];
    s += mp[k] * (v0 * (1.f - fr) * in0 + v1 * fr * in1);
  }
  out[(size_t)row * C + c] = s;
}

// ---------------------------------------------------------------------------
// Host side
// ---------------------------------------------------------------------------
template<int KD, int NT, int EP, int STM>
static inline void gemm(hipStream_t s, const float* A, int lda, const float* W,
                        const float* bias, int Md,
                        const float* other, int ldo,
                        float* out, int ldout, int colbase)
{
  dim3 grid(NLC / 64, (unsigned)((Md + 16 * NT - 1) / (16 * NT)));
  k_gemm<KD, NT, EP, STM><<<grid, 128, 0, s>>>(A, lda, W, bias, Md, other, ldo,
                                               out, ldout, colbase);
}

template<int C>
static inline void run_dcn(hipStream_t stream, void* const* d_in, int G, int Kk, int base,
                           const float* Pin, float* S_V, float* S_F, float* S_O,
                           float* S_M, float* S_S, float* S_A)
{
  auto P = [&](int i) { return (const float*)d_in[i]; };
  const int TPB = 256;
  const int nthr = NLC * C;
  // base: dw.w (+1 dw.b, +2 off.w, +3 off.b, +4 mask.w, +5 mask.b,
  //             +6 in.w, +7 in.b, +8 out.w, +9 out.b)
  gemm<C, 4, EP_BIAS, ST_CL>(stream, Pin, C, P(base + 6), P(base + 7), C,
                             nullptr, 0, S_V, C, 0);                     // value
  k_dwconv<<<(nthr + TPB - 1) / TPB, TPB, 0, stream>>>(
      Pin, C, C, P(base + 0), P(base + 1), Kk, 1, nullptr, 0, S_F, C);   // feat
  gemm<C, 1, EP_BIAS, ST_CL>(stream, S_F, C, P(base + 2), P(base + 3), G * Kk,
                             nullptr, 0, S_O, G * Kk, 0);                // offsets
  gemm<C, 1, EP_BIAS, ST_CL>(stream, S_F, C, P(base + 4), P(base + 5), G * Kk,
                             nullptr, 0, S_M, G * Kk, 0);                // mask logits
  k_softmax<<<(NLC * G + TPB - 1) / TPB, TPB, 0, stream>>>(S_M, G, Kk);
  k_dcn_sample<<<(nthr + TPB - 1) / TPB, TPB, 0, stream>>>(
      S_V, S_O, S_M, C, G, Kk, S_S);
  gemm<C, 4, EP_BIAS, ST_CL>(stream, S_S, C, P(base + 8), P(base + 9), C,
                             nullptr, 0, S_A, C, 0);                     // out proj
}

extern "C" void kernel_launch(void* const* d_in, const int* in_sizes, int n_in,
                              void* d_out, int out_size, void* d_ws, size_t ws_size,
                              hipStream_t stream)
{
  // Flat input indexing (recursive insertion-order flatten of setup_inputs):
  //  0 x
  //  1 norm1.w  2 norm1.b   3 norm2.w  4 norm2.b   5 norm3.w  6 norm3.b
  //  7 a1_pw.w  8 a1_pw.b
  //  dcn1: 9 dw.w 10 dw.b 11 off.w 12 off.b 13 mask.w 14 mask.b 15 in.w 16 in.b 17 out.w 18 out.b
  // 19 v1.w 20 v1.b  21 v11.w 22 v11.b  23 v12.w 24 v12.b  25 conv3_1.w 26 conv3_1.b
  // 27 a2_pw.w 28 a2_pw.b   dcn2: 29..38
  // 39 v2.w 40 v2.b  41 v21.w 42 v21.b  43 v22.w 44 v22.b  45 proj2.w 46 proj2.b
  // 47 conv3_2.w 48 conv3_2.b
  // 49 a3_pw.w 50 a3_pw.b   dcn3: 51..60
  // 61 v3.w 62 v3.b  63 v31.w 64 v31.b  65 v32.w 66 v32.b  67 proj3.w 68 proj3.b
  // 69 conv3_3.w 70 conv3_3.b
  auto P = [&](int i) { return (const float*)d_in[i]; };
  const float* X  = P(0);
  float* OUT = (float*)d_out;

  float* w = (float*)d_ws;
  const size_t SLOT = (size_t)NLC * 256;
  float* S_XN  = w + 0 * SLOT;   // normalized input, CL (NL,256); xs_i = S_XN + i*64
  float* S_CAT = w + 1 * SLOT;   // concat buffer (X1CAT ld128, then X2CAT ld192)
  float* S_NRM = w + 2 * SLOT;   // LN output (X1N ld128, then X2N ld192)
  float* S_P   = w + 3 * SLOT;   // gelu(pw) branch input, reused as U = pw*dcn
  float* S_V   = w + 4 * SLOT;   // dcn value, reused as dw-conv temp
  float* S_F   = w + 5 * SLOT;   // dcn feat,  reused as pw(v12/v22/v32) temp
  float* S_O   = w + 6 * SLOT;   // dcn offsets (NL,G*K)
  float* S_M   = S_O + (size_t)NLC * 128;  // dcn mask (NL,G*K) in same slot
  float* S_S   = w + 7 * SLOT;   // dcn sampled
  float* S_A   = w + 8 * SLOT;   // dcn output (a1/a2/a3), CL

  const int TPB = 256;

  // LN1: (N,256,L) -> channels-last
  k_ln_cf2cl<<<NLC / TPB, TPB, 0, stream>>>(X, P(1), P(2), S_XN);

  // ---------------- branch 1 (C1=64, G=2, K=7) ----------------
  gemm<64, 4, EP_GELU, ST_CL>(stream, S_XN + 0, 256, P(7), P(8), 64,
                              nullptr, 0, S_P, 64, 0);             // gelu(pw(xs0))
  run_dcn<64>(stream, d_in, 2, 7, 9, S_P, S_V, S_F, S_O, S_M, S_S, S_A);
  gemm<64, 4, EP_MUL, ST_CL>(stream, S_XN + 0, 256, P(19), P(20), 64,
                             S_A, 64, S_P, 64, 0);                 // U = pw(xs0,v1)*a1
  gemm<64, 4, EP_BIAS, ST_CL>(stream, S_P, 64, P(21), P(22), 64,
                              nullptr, 0, S_CAT + 64, 128, 0);     // mul1 -> CAT[:,64:]
  gemm<64, 4, EP_BIAS, ST_CL>(stream, S_XN + 64, 256, P(23), P(24), 64,
                              nullptr, 0, S_F, 64, 0);             // pw(xs1,v12)
  k_dwconv<<<(NLC * 64 + TPB - 1) / TPB, TPB, 0, stream>>>(
      S_F, 64, 64, P(25), P(26), 3, 2, S_A, 64, S_CAT, 128);       // x1 = dw3 + a1
  k_ln_cl<<<NLC / TPB, TPB, 0, stream>>>(S_CAT, 128, P(3), P(4), S_NRM); // X1N

  // ---------------- branch 2 (C2=128, G=4, K=9) ----------------
  gemm<128, 4, EP_GELU, ST_CL>(stream, S_NRM, 128, P(27), P(28), 128,
                               nullptr, 0, S_P, 128, 0);
  run_dcn<128>(stream, d_in, 4, 9, 29, S_P, S_V, S_F, S_O, S_M, S_S, S_A);
  gemm<128, 4, EP_MUL, ST_CL>(stream, S_NRM, 128, P(39), P(40), 128,
                              S_A, 128, S_P, 128, 0);              // U = pw(x1,v2)*a2
  gemm<128, 4, EP_BIAS, ST_CL>(stream, S_P, 128, P(41), P(42), 128,
                               nullptr, 0, S_CAT + 64, 192, 0);    // mul2 -> CAT[:,64:192]
  gemm<64, 4, EP_BIAS, ST_CL>(stream, S_XN + 128, 256, P(43), P(44), 64,
                              nullptr, 0, S_F, 64, 0);             // pw(xs2,v22)
  k_dwconv<<<(NLC * 64 + TPB - 1) / TPB, TPB, 0, stream>>>(
      S_F, 64, 64, P(47), P(48), 3, 0, nullptr, 0, S_V, 64);       // dw3
  gemm<128, 4, EP_ADD, ST_CL>(stream, S_A, 128, P(45), P(46), 64,
                              S_V, 64, S_CAT, 192, 0);             // x2 = pw(a2,proj2)+dw
  k_ln_cl<<<NLC / TPB, TPB, 0, stream>>>(S_CAT, 192, P(5), P(6), S_NRM); // X2N

  // ---------------- branch 3 (C3=192, G=6, K=11) ----------------
  gemm<192, 4, EP_GELU, ST_CL>(stream, S_NRM, 192, P(49), P(50), 192,
                               nullptr, 0, S_P, 192, 0);
  run_dcn<192>(stream, d_in, 6, 11, 51, S_P, S_V, S_F, S_O, S_M, S_S, S_A);
  gemm<192, 4, EP_MUL, ST_CL>(stream, S_NRM, 192, P(61), P(62), 192,
                              S_A, 192, S_P, 192, 0);              // U = pw(x2,v3)*a3
  gemm<192, 4, EP_BIAS, ST_CF>(stream, S_P, 192, P(63), P(64), 192,
                               nullptr, 0, OUT, 0, 64);            // mul3 -> out ch 64..255
  gemm<64, 4, EP_BIAS, ST_CL>(stream, S_XN + 192, 256, P(65), P(66), 64,
                              nullptr, 0, S_F, 64, 0);             // pw(xs3,v32)
  k_dwconv<<<(NLC * 64 + TPB - 1) / TPB, TPB, 0, stream>>>(
      S_F, 64, 64, P(69), P(70), 3, 0, nullptr, 0, S_V, 64);       // dw3
  gemm<192, 4, EP_ADD, ST_CF>(stream, S_A, 192, P(67), P(68), 64,
                              S_V, 64, OUT, 0, 0);                 // x3 -> out ch 0..63
}
